// SetAbstraction_89438398972542
// MI455X (gfx1250) — matrix operand order repaired and checked by reading, not compiled, not run on hardware
//
#include <hip/hip_runtime.h>

#define NPC   2048          // points per cloud
#define NCLD  8
#define NTOT  (NPC * NCLD)  // 16384
#define KMAX  32            // max neighbors
#define R2    0.04f         // radius^2
#define FOUT  128

typedef __attribute__((ext_vector_type(16))) _Float16     v16h;
typedef __attribute__((ext_vector_type(8)))  float        v8f;
typedef __attribute__((ext_vector_type(4)))  unsigned int u32x4;

union FragU { u32x4 u[2]; v16h v; };

// A fragment (16x32 f16, ISA layout): lane l holds row M = l%16,
// halves 0..7 -> K = g*8 + h, halves 8..15 -> K = 16 + g*8 + h (g = l/16).
// p points at an LDS matrix with row stride rsH halves; kbase selects the
// 32-wide K chunk.
__device__ __forceinline__ v16h ld_a(const _Float16* p, int rsH, int mrow,
                                     int kbase, int lane) {
  const int l = lane & 15, g = lane >> 4;
  const _Float16* r = p + (mrow + l) * rsH + kbase + g * 8;
  FragU f;
  f.u[0] = *(const u32x4*)(r);
  f.u[1] = *(const u32x4*)(r + 16);
  return f.v;
}

// B fragment (32x16 f16): lane l holds column N = l%16,
// halves 0..7 -> K = g*8 + h, halves 8..15 -> K = 16 + g*8 + h.  Weights are stored TRANSPOSED in LDS
// ([N][K], row stride rsH halves) so this is two 16B reads.
__device__ __forceinline__ v16h ld_b(const _Float16* p, int rsH, int ncol,
                                     int kbase, int lane) {
  const int l = lane & 15, g = lane >> 4;
  const _Float16* r = p + (ncol + l) * rsH + kbase + g * 8;
  FragU f;
  f.u[0] = *(const u32x4*)(r);
  f.u[1] = *(const u32x4*)(r + 16);
  return f.v;
}

#define WMMA_F16(a, b, c) \
  __builtin_amdgcn_wmma_f32_16x16x32_f16(false, (a), false, (b), (short)0, (c), false, false)

// ---------------------------------------------------------------------------
// Kernel 1: radius ball query, first KMAX in index order (self always valid).
// One thread per query point; pos is L2-resident so the scan is cheap.
// ---------------------------------------------------------------------------
__global__ __launch_bounds__(256) void nbr_kernel(const float* __restrict__ pos,
                                                  int* __restrict__ nbr,
                                                  int* __restrict__ cnt) {
  const int q = blockIdx.x * blockDim.x + threadIdx.x;
  if (q >= NTOT) return;
  const int base = (q / NPC) * NPC;
  const float px = pos[q * 3 + 0], py = pos[q * 3 + 1], pz = pos[q * 3 + 2];
  int c = 0;
  for (int j = 0; j < NPC && c < KMAX; ++j) {
    const int gidx = base + j;
    const float dx = pos[gidx * 3 + 0] - px;
    const float dy = pos[gidx * 3 + 1] - py;
    const float dz = pos[gidx * 3 + 2] - pz;
    if (dx * dx + dy * dy + dz * dz <= R2) nbr[q * KMAX + (c++)] = gidx;
  }
  cnt[q] = c;
}

// ---------------------------------------------------------------------------
// Kernel 2: gather -> 3-layer MLP (WMMA f16, f32 accum) -> masked max.
// 4 waves/block; each wave owns one query (32 neighbor rows = 2 M-tiles).
// ---------------------------------------------------------------------------
__global__ __launch_bounds__(128)
void mlp_kernel(const float* __restrict__ x, const float* __restrict__ pos,
                const int* __restrict__ nbr, const int* __restrict__ cnt,
                const float* __restrict__ W1, const float* __restrict__ B1,
                const float* __restrict__ W2, const float* __restrict__ B2,
                const float* __restrict__ W3, const float* __restrict__ B3,
                float* __restrict__ out) {
  __shared__ alignas(16) _Float16 sW1[64 * 32];    // [N=64][K=32] (K padded 6->32)
  __shared__ alignas(16) _Float16 sW2[64 * 64];    // [N=64][K=64]
  __shared__ alignas(16) _Float16 sW3[128 * 64];   // [N=128][K=64]
  __shared__ float sB1[64], sB2[64], sB3[128];
  __shared__ alignas(16) _Float16 sA[4][32 * 32];  // per-wave input stage
  __shared__ alignas(16) _Float16 sH[4][32 * 64];  // per-wave activation stage

  const int tid = threadIdx.x;

  // Cooperative weight convert f32 -> f16, transposed [N][K].
  for (int i = tid; i < 64 * 32; i += 128) {
    const int n = i >> 5, k = i & 31;
    sW1[i] = (k < 6) ? (_Float16)W1[k * 64 + n] : (_Float16)0.f;
  }
  for (int i = tid; i < 64 * 64; i += 128) {
    const int n = i >> 6, k = i & 63;
    sW2[i] = (_Float16)W2[k * 64 + n];
  }
  for (int i = tid; i < 128 * 64; i += 128) {
    const int n = i >> 6, k = i & 63;
    sW3[i] = (_Float16)W3[k * 128 + n];
  }
  if (tid < 64) { sB1[tid] = B1[tid]; sB2[tid] = B2[tid]; }
  if (tid < 128) sB3[tid] = B3[tid];
  __syncthreads();

  const int wid = tid >> 5, lane = tid & 31;
  const int g = lane >> 4, l15 = lane & 15;
  const int q = blockIdx.x * 4 + wid;
  const int cq = cnt[q];
  const float pix = pos[q * 3 + 0], piy = pos[q * 3 + 1], piz = pos[q * 3 + 2];

  // Build A-stage: lane = neighbor slot; row = [x_j(3), pos_j - pos_i(3), 0...].
  {
    float f0 = 0.f, f1 = 0.f, f2 = 0.f, f3 = 0.f, f4 = 0.f, f5 = 0.f;
    if (lane < cq) {
      const int j = nbr[q * KMAX + lane];
      f0 = x[j * 3 + 0]; f1 = x[j * 3 + 1]; f2 = x[j * 3 + 2];
      f3 = pos[j * 3 + 0] - pix; f4 = pos[j * 3 + 1] - piy; f5 = pos[j * 3 + 2] - piz;
    }
    union { _Float16 h[32]; u32x4 u4[4]; } rowu;
#pragma unroll
    for (int i = 0; i < 32; ++i) rowu.h[i] = (_Float16)0.f;
    rowu.h[0] = (_Float16)f0; rowu.h[1] = (_Float16)f1; rowu.h[2] = (_Float16)f2;
    rowu.h[3] = (_Float16)f3; rowu.h[4] = (_Float16)f4; rowu.h[5] = (_Float16)f5;
    u32x4* dst = (u32x4*)&sA[wid][lane * 32];
    dst[0] = rowu.u4[0]; dst[1] = rowu.u4[1]; dst[2] = rowu.u4[2]; dst[3] = rowu.u4[3];
  }
  __syncthreads();

  // ---- Layer 1: [32,32(pad6)] x [32,64] -> relu -> sH --------------------
#pragma unroll
  for (int mt = 0; mt < 2; ++mt) {
    const v16h a = ld_a(sA[wid], 32, mt * 16, 0, lane);
#pragma unroll
    for (int nt = 0; nt < 4; ++nt) {
      const v16h b = ld_b(sW1, 32, nt * 16, 0, lane);
      v8f c = {};
      c = WMMA_F16(a, b, c);
      const float bias = sB1[nt * 16 + l15];
#pragma unroll
      for (int r = 0; r < 8; ++r) {
        float v = c[r] + bias;
        v = v > 0.f ? v : 0.f;
        sH[wid][(mt * 16 + g * 8 + r) * 64 + nt * 16 + l15] = (_Float16)v;
      }
    }
  }
  __syncthreads();

  // ---- Layer 2: [32,64] x [64,64] -> relu -> sH (in place; per-M-tile A is
  // fully loaded to registers before that tile's rows are overwritten, and
  // same-wave LDS ops are in-order) ---------------------------------------
#pragma unroll
  for (int mt = 0; mt < 2; ++mt) {
    const v16h a0 = ld_a(sH[wid], 64, mt * 16, 0, lane);
    const v16h a1 = ld_a(sH[wid], 64, mt * 16, 32, lane);
#pragma unroll
    for (int nt = 0; nt < 4; ++nt) {
      const v16h b0 = ld_b(sW2, 64, nt * 16, 0, lane);
      const v16h b1 = ld_b(sW2, 64, nt * 16, 32, lane);
      v8f c = {};
      c = WMMA_F16(a0, b0, c);
      c = WMMA_F16(a1, b1, c);
      const float bias = sB2[nt * 16 + l15];
#pragma unroll
      for (int r = 0; r < 8; ++r) {
        float v = c[r] + bias;
        v = v > 0.f ? v : 0.f;
        sH[wid][(mt * 16 + g * 8 + r) * 64 + nt * 16 + l15] = (_Float16)v;
      }
    }
  }
  __syncthreads();

  // ---- Layer 3: [32,64] x [64,128] + bias, mask invalid rows, max over 32 -
  const float NEG = -__builtin_inff();
  float colmax[8];
#pragma unroll
  for (int nt = 0; nt < 8; ++nt) colmax[nt] = NEG;

#pragma unroll
  for (int mt = 0; mt < 2; ++mt) {
    const v16h a0 = ld_a(sH[wid], 64, mt * 16, 0, lane);
    const v16h a1 = ld_a(sH[wid], 64, mt * 16, 32, lane);
#pragma unroll
    for (int nt = 0; nt < 8; ++nt) {
      const v16h b0 = ld_b(sW3, 64, nt * 16, 0, lane);
      const v16h b1 = ld_b(sW3, 64, nt * 16, 32, lane);
      v8f c = {};
      c = WMMA_F16(a0, b0, c);
      c = WMMA_F16(a1, b1, c);
      const float bias = sB3[nt * 16 + l15];
      float m = NEG;
#pragma unroll
      for (int r = 0; r < 8; ++r) {
        const int kslot = mt * 16 + g * 8 + r;  // C layout: M = 8*g + r
        const float v = (kslot < cq) ? (c[r] + bias) : NEG;
        m = fmaxf(m, v);
      }
      m = fmaxf(m, __shfl_xor(m, 16, 32));  // fold rows 0-7 with 8-15
      colmax[nt] = fmaxf(colmax[nt], m);
    }
  }

  if (lane < 16) {
#pragma unroll
    for (int nt = 0; nt < 8; ++nt)
      out[q * FOUT + nt * 16 + lane] = colmax[nt];
  }
}

// ---------------------------------------------------------------------------
// Kernel 3: tuple tail — copy pos, bit-copy batch.
// ---------------------------------------------------------------------------
__global__ __launch_bounds__(256) void tail_kernel(const float* __restrict__ pos,
                                                   const int* __restrict__ batch,
                                                   float* __restrict__ out) {
  const int i = blockIdx.x * blockDim.x + threadIdx.x;
  if (i < NTOT * 3) out[NTOT * FOUT + i] = pos[i];
  if (i < NTOT) ((int*)out)[NTOT * FOUT + NTOT * 3 + i] = batch[i];
}

extern "C" void kernel_launch(void* const* d_in, const int* in_sizes, int n_in,
                              void* d_out, int out_size, void* d_ws, size_t ws_size,
                              hipStream_t stream) {
  const float* x   = (const float*)d_in[0];
  const float* pos = (const float*)d_in[1];
  const int* batch = (const int*)d_in[2];
  const float* W1  = (const float*)d_in[3];
  const float* b1  = (const float*)d_in[4];
  const float* W2  = (const float*)d_in[5];
  const float* b2  = (const float*)d_in[6];
  const float* W3  = (const float*)d_in[7];
  const float* b3  = (const float*)d_in[8];
  float* out = (float*)d_out;

  int* nbr = (int*)d_ws;            // NTOT*KMAX ints
  int* cnt = nbr + NTOT * KMAX;     // NTOT ints

  nbr_kernel<<<NTOT / 256, 256, 0, stream>>>(pos, nbr, cnt);
  mlp_kernel<<<NTOT / 4, 128, 0, stream>>>(x, pos, nbr, cnt,
                                           W1, b1, W2, b2, W3, b3, out);
  tail_kernel<<<(NTOT * 3 + 255) / 256, 256, 0, stream>>>(pos, batch, out);
}
